// CorrelationLayer_62809601737101
// MI455X (gfx1250) — matrix operand order repaired and checked by reading, not compiled, not run on hardware
//
#include <hip/hip_runtime.h>
#include <hip/hip_bf16.h>
#include <stdint.h>

// ---------------------------------------------------------------------------
// CorrelationLayer fused pipeline for MI455X (gfx1250, wave32, WMMA bf16).
//   y1 = (W1 @ ker[b]^T) @ feat * rinv(hw) + b1   (corr never materialized)
// B-operands stored pixel-major K-pair-packed so GEMM LDS staging is a
// contiguous 128b copy (async global->LDS when available); every WMMA
// fragment is 2x ds_load_b128 / global_load_b128.
// ---------------------------------------------------------------------------

typedef __attribute__((ext_vector_type(16))) __bf16   v16bf;
typedef __attribute__((ext_vector_type(8)))  float    v8f;
typedef __attribute__((ext_vector_type(4)))  uint32_t v4u;
typedef __attribute__((ext_vector_type(4)))  int      v4i;

union FragU { uint32_t u[8]; v4u q[2]; v16bf v; };

#define BATCH 16
#define CH    256
#define CHP   128          // channel pairs
#define IMH   96
#define IMW   96
#define HW    (IMH*IMW)    // 9216
#define FCO   256

#if defined(__AMDGCN__) && __has_builtin(__builtin_amdgcn_global_load_async_to_lds_b128)
#define USE_ASYNC_LDS 1
#else
#define USE_ASYNC_LDS 0
#endif

__device__ __forceinline__ uint16_t f2bf(float x){
  uint32_t u = __builtin_bit_cast(uint32_t, x);
  uint32_t r = u + 0x7FFFu + ((u >> 16) & 1u);   // round-to-nearest-even
  return (uint16_t)(r >> 16);
}
__device__ __forceinline__ float bf2f(uint16_t b){
  return __builtin_bit_cast(float, (uint32_t)b << 16);
}
__device__ __forceinline__ uint32_t pack2(uint16_t lo, uint16_t hi){
  return (uint32_t)lo | ((uint32_t)hi << 16);
}
__device__ __forceinline__ v8f wmma_bf16(const FragU& a, const FragU& b, v8f c){
  return __builtin_amdgcn_wmma_f32_16x16x32_bf16(false, a.v, false, b.v,
                                                 (short)0, c, false, false);
}
// A fragment: row-major bf16 matrix viewed as dwords (K-pairs).
__device__ __forceinline__ void load_a_frag(FragU& a, const uint32_t* row_k0, int g){
  a.q[0] = *(const v4u*)(row_k0 + g*8);
  a.q[1] = *(const v4u*)(row_k0 + g*8 + 4);
}
// Stage 16 bytes global -> LDS (async DMA when the toolchain exposes it).
__device__ __forceinline__ void stage_b128(const uint32_t* gsrc, uint32_t* ldst){
#if USE_ASYNC_LDS
  __builtin_amdgcn_global_load_async_to_lds_b128(
      (__attribute__((address_space(1))) v4i*)gsrc,
      (__attribute__((address_space(3))) v4i*)ldst, 0, 0);
#else
  *(v4u*)ldst = *(const v4u*)gsrc;
#endif
}
__device__ __forceinline__ void wait_async_stage(){
#if USE_ASYNC_LDS
#if __has_builtin(__builtin_amdgcn_s_wait_asynccnt)
  __builtin_amdgcn_s_wait_asynccnt(0);
#else
  asm volatile("s_wait_asynccnt 0x0" ::: "memory");
#endif
#endif
}

// --------------------------- weight conversion -----------------------------
// W1bf: [o][p] bf16.  W2t: tap-major [tap][o][c] bf16 (pairs contiguous in c)
__global__ __launch_bounds__(256) void k_convert_weights(
    const float* __restrict__ W1, const float* __restrict__ W2,
    uint16_t* __restrict__ W1bf, uint16_t* __restrict__ W2t){
  int i = blockIdx.x * 256 + threadIdx.x;
  if (i < FCO*CH) W1bf[i] = f2bf(W1[i]);
  if (i < FCO*CH*9){
    int tap = i >> 16, rem = i & 65535;          // dst = tap*65536 + (o*256+c)
    W2t[i] = f2bf(W2[rem*9 + tap]);
  }
}

// ------------- adaptive 16x16 pool -> ker^T, c-major p-pair packed ---------
// kerTc[b][c][pp] = pack(mean block 2pp, mean block 2pp+1) of feat[b,c]
__global__ __launch_bounds__(256) void k_pool16(
    const float* __restrict__ feat, uint32_t* __restrict__ kerTc){
  int idx = blockIdx.x * 256 + threadIdx.x;      // b*32768 + pp*256 + c
  int b = idx >> 15, pp = (idx >> 8) & 127, c = idx & 255;
  const float* base = feat + ((size_t)(b*CH + c))*HW;
  float s[2];
  #pragma unroll
  for (int t = 0; t < 2; ++t){
    int p = 2*pp + t, py = p >> 4, px = p & 15;
    const float* src = base + (py*6)*IMW + px*6;
    float acc = 0.f;
    #pragma unroll
    for (int i = 0; i < 6; ++i)
      #pragma unroll
      for (int j = 0; j < 6; ++j) acc += src[i*IMW + j];
    s[t] = acc * (1.f/36.f);
  }
  kerTc[((size_t)(b*CH + c))*CHP + pp] = pack2(f2bf(s[0]), f2bf(s[1]));
}

// -- per-pixel L2 norm reciprocal + pixel-major channel-pair-packed copy ----
__global__ __launch_bounds__(256) void k_norm_cvt(
    const float* __restrict__ feat, uint32_t* __restrict__ featpk,
    float* __restrict__ rinv){
  int b  = blockIdx.y;
  int hw = blockIdx.x * 256 + threadIdx.x;
  const float* src = feat + (size_t)b*CH*HW + hw;
  uint32_t* dst = featpk + ((size_t)(b*HW) + hw)*CHP;
  float s = 0.f;
  uint32_t buf[4];
  for (int cp = 0; cp < CHP; ++cp){
    float v0 = src[(size_t)(2*cp)*HW];
    float v1 = src[(size_t)(2*cp+1)*HW];
    s += v0*v0 + v1*v1;
    buf[cp & 3] = pack2(f2bf(v0), f2bf(v1));
    if ((cp & 3) == 3) *(v4u*)(dst + cp - 3) = *(const v4u*)buf;
  }
  rinv[b*HW + hw] = 1.f / fmaxf(sqrtf(s), 1e-12f);
}

// ---------------- M[b] = W1 @ ker[b]^T   (16 GEMMs, 256x256x256) -----------
__global__ __launch_bounds__(256) void k_gemm_M(
    const uint16_t* __restrict__ W1bf, const uint32_t* __restrict__ kerTc,
    uint16_t* __restrict__ Mbf){
  __shared__ uint32_t bs32[64*20];               // [n][kpair], pad 20
  int b = blockIdx.y;
  int n0 = blockIdx.x * 64;                      // c-tile
  int oBase = blockIdx.z * 128 + (threadIdx.x >> 5) * 16;
  int lane = threadIdx.x & 31, g = lane >> 4, nl = lane & 15;
  const uint32_t* W1w = (const uint32_t*)W1bf;   // [o][128 pairs]
  v8f acc[4] = {};
  for (int k0 = 0; k0 < CH; k0 += 32){
    { // stage: row n = 16 contiguous global dwords -> contiguous LDS
      int n = threadIdx.x >> 2, part = (threadIdx.x & 3) * 4;
      stage_b128(kerTc + ((size_t)(b*CH + n0 + n))*CHP + (k0>>1) + part,
                 &bs32[n*20 + part]);
    }
    wait_async_stage();
    __syncthreads();
    FragU a;
    load_a_frag(a, W1w + (size_t)(oBase + nl)*128 + (k0>>1), g);
    #pragma unroll
    for (int t = 0; t < 4; ++t){
      FragU bb;
      bb.q[0] = *(const v4u*)&bs32[(t*16 + nl)*20 + g*8];
      bb.q[1] = *(const v4u*)&bs32[(t*16 + nl)*20 + g*8 + 4];
      acc[t] = wmma_bf16(a, bb, acc[t]);
    }
    __syncthreads();
  }
  #pragma unroll
  for (int t = 0; t < 4; ++t)
    #pragma unroll
    for (int r = 0; r < 8; ++r){
      int orow = oBase + g*8 + r;
      Mbf[((size_t)b << 16) + (size_t)orow*CH + n0 + t*16 + nl] = f2bf(acc[t][r]);
    }
}

// ------ y1[b] = M[b] @ feat[b] * rinv + b1  (channel-pair-major out) -------
__global__ __launch_bounds__(256) void k_gemm_y1(
    const uint16_t* __restrict__ Mbf, const uint32_t* __restrict__ featpk,
    const float* __restrict__ rinv, const float* __restrict__ b1,
    uint32_t* __restrict__ y1pk){
  __shared__ uint32_t bs32[64*20];
  int b = blockIdx.y;
  int n0 = blockIdx.x * 64;                      // hw-tile
  int oBase = blockIdx.z * 128 + (threadIdx.x >> 5) * 16;
  int lane = threadIdx.x & 31, g = lane >> 4, nl = lane & 15;
  const uint32_t* Mw = (const uint32_t*)Mbf + ((size_t)b << 15); // [o][128]
  const uint32_t* X  = featpk + (size_t)(b*HW)*CHP;              // [hw][cp]
  v8f acc[4] = {};
  for (int k0 = 0; k0 < CH; k0 += 32){
    { // stage: pixel row n -> 16 contiguous dwords
      int n = threadIdx.x >> 2, part = (threadIdx.x & 3) * 4;
      stage_b128(X + (size_t)(n0 + n)*CHP + (k0>>1) + part,
                 &bs32[n*20 + part]);
    }
    wait_async_stage();
    __syncthreads();
    FragU a;
    load_a_frag(a, Mw + (size_t)(oBase + nl)*128 + (k0>>1), g);
    #pragma unroll
    for (int t = 0; t < 4; ++t){
      FragU bb;
      bb.q[0] = *(const v4u*)&bs32[(t*16 + nl)*20 + g*8];
      bb.q[1] = *(const v4u*)&bs32[(t*16 + nl)*20 + g*8 + 4];
      acc[t] = wmma_bf16(a, bb, acc[t]);
    }
    __syncthreads();
  }
  #pragma unroll
  for (int t = 0; t < 4; ++t){
    int col = n0 + t*16 + nl;
    float rv = rinv[b*HW + col];
    #pragma unroll
    for (int r2 = 0; r2 < 4; ++r2){
      int orow = oBase + g*8 + 2*r2;             // even: pair (orow, orow+1)
      float y0 = acc[t][2*r2]   * rv + b1[orow];
      float y1 = acc[t][2*r2+1] * rv + b1[orow+1];
      y1pk[((size_t)(b*CHP + (orow >> 1)))*HW + col] = pack2(f2bf(y0), f2bf(y1));
    }
  }
}

// --------------- InstanceNorm stats (mean, rstd) per channel pair ----------
__global__ __launch_bounds__(256) void k_instnorm_stats(
    const uint32_t* __restrict__ y1pk, float* __restrict__ stats){
  __shared__ float red[4][256];
  int b = blockIdx.y, cp = blockIdx.x;
  const uint32_t* p = y1pk + ((size_t)(b*CHP + cp))*HW;
  float a0 = 0.f, q0 = 0.f, a1 = 0.f, q1 = 0.f;
  for (int i = threadIdx.x; i < HW; i += 256){
    uint32_t w = p[i];
    float v0 = bf2f((uint16_t)w), v1 = bf2f((uint16_t)(w >> 16));
    a0 += v0; q0 += v0*v0; a1 += v1; q1 += v1*v1;
  }
  red[0][threadIdx.x] = a0; red[1][threadIdx.x] = q0;
  red[2][threadIdx.x] = a1; red[3][threadIdx.x] = q1;
  __syncthreads();
  for (int off = 128; off > 0; off >>= 1){
    if ((int)threadIdx.x < off)
      #pragma unroll
      for (int j = 0; j < 4; ++j)
        red[j][threadIdx.x] += red[j][threadIdx.x + off];
    __syncthreads();
  }
  if (threadIdx.x == 0){
    #pragma unroll
    for (int t = 0; t < 2; ++t){
      float mu  = red[2*t][0] * (1.f/(float)HW);
      float var = red[2*t+1][0] * (1.f/(float)HW) - mu*mu;
      int c = 2*cp + t;
      stats[(b*CH + c)*2]     = mu;
      stats[(b*CH + c)*2 + 1] = rsqrtf(fmaxf(var, 0.f) + 1e-5f);
    }
  }
}

// --- 3x3 conv, full-row tile (96 px); IN+ReLU fused into LDS staging -------
__global__ __launch_bounds__(256) void k_conv3x3(
    const uint32_t* __restrict__ y1pk, const uint16_t* __restrict__ W2t,
    const float* __restrict__ stats, const float* __restrict__ b2,
    float* __restrict__ out){
  __shared__ uint32_t sxp[98*3*20];              // [(col*3+row)*20 + kpair]
  __shared__ float smu[CH], srs[CH];
  int b  = blockIdx.y;
  int h  = blockIdx.x;                            // one image row per WG
  int oBase = blockIdx.z * 128 + (threadIdx.x >> 5) * 16;
  int lane = threadIdx.x & 31, g = lane >> 4, nl = lane & 15;
  if (threadIdx.x < CH){
    smu[threadIdx.x] = stats[(b*CH + threadIdx.x)*2];
    srs[threadIdx.x] = stats[(b*CH + threadIdx.x)*2 + 1];
  }
  __syncthreads();
  const uint32_t* Y = y1pk + (size_t)b*CHP*HW;
  const uint32_t* W2w = (const uint32_t*)W2t;    // [tap][o][128 pairs]
  v8f acc[6] = {};
  for (int c0 = 0; c0 < CH; c0 += 32){
    for (int i = threadIdx.x; i < 16*3*98; i += 256){
      int kp = i / 294, sp = i % 294, rr = sp / 98, col = sp % 98;
      int hs = h + rr - 1, ws = col - 1;
      int cp = (c0 >> 1) + kp;
      uint32_t pkd = 0;
      if ((unsigned)hs < (unsigned)IMH && (unsigned)ws < (unsigned)IMW){
        uint32_t w = Y[(size_t)cp*HW + hs*IMW + ws];
        int c = 2*cp;
        float x0 = fmaxf((bf2f((uint16_t)w)         - smu[c])   * srs[c],   0.f);
        float x1 = fmaxf((bf2f((uint16_t)(w >> 16)) - smu[c+1]) * srs[c+1], 0.f);
        pkd = pack2(f2bf(x0), f2bf(x1));
      }
      sxp[(col*3 + rr)*20 + kp] = pkd;
    }
    __syncthreads();
    #pragma unroll
    for (int dy = 0; dy < 3; ++dy)
      #pragma unroll
      for (int dx = 0; dx < 3; ++dx){
        int tap = dy*3 + dx;
        FragU a;
        load_a_frag(a, W2w + (size_t)tap*32768 + (size_t)(oBase + nl)*128 + (c0>>1), g);
        #pragma unroll
        for (int t = 0; t < 6; ++t){
          FragU bb;
          int cl = (t*16 + nl + dx)*3 + dy;
          bb.q[0] = *(const v4u*)&sxp[cl*20 + g*8];
          bb.q[1] = *(const v4u*)&sxp[cl*20 + g*8 + 4];
          acc[t] = wmma_bf16(a, bb, acc[t]);
        }
      }
    __syncthreads();
  }
  #pragma unroll
  for (int r = 0; r < 8; ++r){
    int orow = oBase + g*8 + r;
    float bias = b2[orow];
    size_t o_off = ((size_t)(b*CH + orow))*HW + (size_t)h*IMW;
    #pragma unroll
    for (int t = 0; t < 6; ++t)
      out[o_off + t*16 + nl] = acc[t][r] + bias;
  }
}

// ------------------------------- launcher ----------------------------------
extern "C" void kernel_launch(void* const* d_in, const int* in_sizes, int n_in,
                              void* d_out, int out_size, void* d_ws, size_t ws_size,
                              hipStream_t stream){
  const float* F_ir  = (const float*)d_in[0];
  const float* F_vis = (const float*)d_in[1];
  const float* W1    = (const float*)d_in[2];
  const float* b1    = (const float*)d_in[3];
  const float* W2    = (const float*)d_in[4];
  const float* b2    = (const float*)d_in[5];
  float* out = (float*)d_out;

  char* ws = (char*)d_ws;
  size_t off = 0;
  auto alloc = [&](size_t bytes) -> void* {
    void* p = ws + off;
    off += (bytes + 255) & ~(size_t)255;
    return p;
  };
  uint32_t* featpk = (uint32_t*)alloc(sizeof(uint32_t)*(size_t)BATCH*HW*CHP);
  uint32_t* y1pk   = (uint32_t*)alloc(sizeof(uint32_t)*(size_t)BATCH*CHP*HW);
  uint32_t* kerTc  = (uint32_t*)alloc(sizeof(uint32_t)*(size_t)BATCH*CH*CHP);
  uint16_t* Mbf    = (uint16_t*)alloc(sizeof(uint16_t)*(size_t)BATCH*FCO*CH);
  uint16_t* W1bf   = (uint16_t*)alloc(sizeof(uint16_t)*FCO*CH);
  uint16_t* W2t    = (uint16_t*)alloc(sizeof(uint16_t)*9*FCO*CH);
  float*    rinv   = (float*)alloc(sizeof(float)*(size_t)BATCH*HW);
  float*    stats  = (float*)alloc(sizeof(float)*(size_t)BATCH*CH*2);

  k_convert_weights<<<2304, 256, 0, stream>>>(W1, W2, W1bf, W2t);

  const float* feats[2] = { F_ir, F_vis };     // output order: IR then VIS
  for (int s = 0; s < 2; ++s){
    const float* feat = feats[s];
    float* outs = out + (size_t)s*BATCH*FCO*HW;
    k_pool16<<<2048, 256, 0, stream>>>(feat, kerTc);
    k_norm_cvt<<<dim3(HW/256, BATCH), 256, 0, stream>>>(feat, featpk, rinv);
    k_gemm_M<<<dim3(4, BATCH, 2), 256, 0, stream>>>(W1bf, kerTc, Mbf);
    k_gemm_y1<<<dim3(HW/64, BATCH, 2), 256, 0, stream>>>(Mbf, featpk, rinv, b1, y1pk);
    k_instnorm_stats<<<dim3(CHP, BATCH), 256, 0, stream>>>(y1pk, stats);
    k_conv3x3<<<dim3(IMH, BATCH, 2), 256, 0, stream>>>(y1pk, W2t, stats, b2, outs);
  }
}
